// KGATConv_11269994185390
// MI455X (gfx1250) — matrix operand order repaired and checked by reading, not compiled, not run on hardware
//
#include <hip/hip_runtime.h>
#include <hip/hip_bf16.h>

// ---------------------------------------------------------------------------
// KGAT bi-interaction conv for gfx1250 (MI455X)
//   hn[v]  = sum_{e: dst(e)=v} nfeat[src(e)] * w_e          (atomic scatter)
//   out    = leaky_relu((nfeat .* hn) @ W^T)                 (bf16 WMMA GEMM)
// ---------------------------------------------------------------------------

#define D_FEAT 128

typedef __attribute__((ext_vector_type(16))) __bf16       v16bf;
typedef __attribute__((ext_vector_type(8)))  float        v8f;
typedef __attribute__((ext_vector_type(8)))  unsigned int v8u;

// Fused (x0*y0, x1*y1) -> packed bf16 dword via V_FMA_MIXLO/MIXHI_BF16
// (CDNA5 VOP3P opcodes 62/63: D[15:0]/D[31:16].bf16 = cvt(fma(f32,f32,f32))).
// "=&v": mixlo writes dest before mixhi reads its inputs -> early clobber.
__device__ __forceinline__ unsigned int mul_pack_bf16(float x0, float y0,
                                                      float x1, float y1) {
  unsigned int d;
  asm("v_fma_mixlo_bf16 %0, %1, %2, 0\n\t"
      "v_fma_mixhi_bf16 %0, %3, %4, 0"
      : "=&v"(d)
      : "v"(x0), "v"(y0), "v"(x1), "v"(y1));
  return d;
}

// Pure convert: fma(x, 1.0, 0) -> bf16, packed pair.
__device__ __forceinline__ unsigned int cvt_pack_bf16(float lo, float hi) {
  unsigned int d;
  asm("v_fma_mixlo_bf16 %0, %1, 1.0, 0\n\t"
      "v_fma_mixhi_bf16 %0, %2, 1.0, 0"
      : "=&v"(d)
      : "v"(lo), "v"(hi));
  return d;
}

// leaky_relu(x) == max(x, 0.01*x) for slope in (0,1). Direct v_max_num_f32
// (IEEE maximumNumber in HW) avoids LLVM's sNaN canonicalize (max(x,x)) ops.
__device__ __forceinline__ float leaky01(float x) {
  float y;
  asm("v_mul_f32_e32 %0, 0x3c23d70a, %1\n\t"   // y = 0.01f * x
      "v_max_num_f32_e32 %0, %1, %0"            // y = max(x, y)
      : "=&v"(y)
      : "v"(x));
  return y;
}

// --- Kernel 1: convert W (f32, row-major [j][k]) to packed bf16 pairs -------
__global__ __launch_bounds__(256) void kgat_prep_w(
    const float* __restrict__ W, unsigned int* __restrict__ Wbf) {
  int i = blockIdx.x * blockDim.x + threadIdx.x;           // 128*64 packed uints
  if (i < D_FEAT * (D_FEAT / 2)) {
    float lo = W[2 * i + 0];
    float hi = W[2 * i + 1];
    Wbf[i] = cvt_pack_bf16(lo, hi);
  }
}

// --- Kernel 2: edge scatter-add. One wave32 per edge; lane holds float4. ----
__global__ __launch_bounds__(256) void kgat_scatter(
    const float* __restrict__ nfeat, const int* __restrict__ esrc,
    const int* __restrict__ edst, const float* __restrict__ ew,
    float* __restrict__ hn, int E) {
  long gid = (long)blockIdx.x * blockDim.x + threadIdx.x;
  int e = (int)(gid >> 5);
  if (e >= E) return;
  int lane = (int)gid & 31;

  int   src = esrc[e];
  int   dst = edst[e];
  float w   = ew[e];

  // 32 lanes * float4 == one contiguous 512B row: fully coalesced gather.
  float4 x = reinterpret_cast<const float4*>(nfeat + (size_t)src * D_FEAT)[lane];
  float* pd = hn + (size_t)dst * D_FEAT + lane * 4;

  // hn fits in the 192MB L2: force the HW L2 FP atomic (no CAS loop).
  asm volatile(
      "global_atomic_add_f32 %0, %1, off\n\t"
      "global_atomic_add_f32 %0, %2, off offset:4\n\t"
      "global_atomic_add_f32 %0, %3, off offset:8\n\t"
      "global_atomic_add_f32 %0, %4, off offset:12"
      :
      : "v"(pd), "v"(x.x * w), "v"(x.y * w), "v"(x.z * w), "v"(x.w * w)
      : "memory");
}

// --- Kernel 3: fused (nfeat .* hn) @ W^T + leaky_relu via WMMA bf16 ---------
// One wave per 16-node tile; sweeps 8 column tiles x 4 K-steps = 32 WMMAs.
__global__ __launch_bounds__(256) void kgat_gemm(
    const float* __restrict__ nfeat, const float* __restrict__ hn,
    const unsigned int* __restrict__ Wbf, float* __restrict__ out, int N) {
  const int lane   = threadIdx.x & 31;
  const int wave   = threadIdx.x >> 5;
  const int ntiles = (N + 15) >> 4;
  const int tile   = blockIdx.x * 8 + wave;
  if (tile >= ntiles) return;          // wave-uniform: EXEC all-1s for WMMA

  const int half = lane >> 4;          // lane group (0: lanes 0-15, 1: 16-31)
  const int m    = lane & 15;
  const bool full_tile = (tile * 16 + 16) <= N;   // wave-uniform

  int row = tile * 16 + m;             // node index held by this lane (A rows)
  if (row >= N) row = N - 1;           // clamp loads; stores predicated below

  const float2* pn = reinterpret_cast<const float2*>(nfeat + (size_t)row * D_FEAT);
  const float2* ph = reinterpret_cast<const float2*>(hn    + (size_t)row * D_FEAT);

  // Build A fragments (A = nfeat .* hn, bf16) for the 4 K-blocks of 32.
  // A k order used here (contiguous, same as B): K = 16*half + 2v (+1 packed).
  v8u apk[4];
#pragma unroll
  for (int kb = 0; kb < 4; ++kb) {
#pragma unroll
    for (int v = 0; v < 8; ++v) {
      int kbase = 16 * half + 2 * v;
      int k     = kb * 32 + kbase;
      float2 x  = pn[k >> 1];
      float2 y  = ph[k >> 1];
      apk[kb][v] = mul_pack_bf16(x.x, y.x, x.y, y.y);
    }
  }

#pragma unroll
  for (int c = 0; c < 8; ++c) {        // 8 column tiles of 16 output features
    v8f acc = {};
    const int j = c * 16 + m;          // W row (output feature) for this lane
#pragma unroll
    for (int kb = 0; kb < 4; ++kb) {
      // B k order used here (contiguous, same as A): lane = N, K = kb*32 + 16*half + 2v (+1 packed).
      const unsigned int* pw = Wbf + (size_t)j * (D_FEAT / 2) + kb * 16 + 8 * half;
      v8u bpk;
#pragma unroll
      for (int v = 0; v < 8; ++v) bpk[v] = pw[v];

      acc = __builtin_amdgcn_wmma_f32_16x16x32_bf16(
          false, __builtin_bit_cast(v16bf, apk[kb]),
          false, __builtin_bit_cast(v16bf, bpk),
          (short)0, acc, false, false);
    }
    // C/D layout: VGPR r -> row M = r + 8*half, col N = m.
    float* po = out + (size_t)(tile * 16 + 8 * half) * D_FEAT + c * 16 + m;
    if (full_tile) {
#pragma unroll
      for (int r = 0; r < 8; ++r) {
        po[(size_t)r * D_FEAT] = leaky01(acc[r]);
      }
    } else {
#pragma unroll
      for (int r = 0; r < 8; ++r) {
        int orow = tile * 16 + r + 8 * half;
        if (orow < N) {
          out[(size_t)orow * D_FEAT + c * 16 + m] = leaky01(acc[r]);
        }
      }
    }
  }
}

extern "C" void kernel_launch(void* const* d_in, const int* in_sizes, int n_in,
                              void* d_out, int out_size, void* d_ws, size_t ws_size,
                              hipStream_t stream) {
  const float* nfeat = (const float*)d_in[0];
  const int*   esrc  = (const int*)d_in[1];
  const int*   edst  = (const int*)d_in[2];
  const float* ew    = (const float*)d_in[3];
  const float* W     = (const float*)d_in[4];
  float*       out   = (float*)d_out;

  const int N = in_sizes[0] / D_FEAT;   // 100000
  const int E = in_sizes[1];            // 1600000

  float*        hn  = (float*)d_ws;
  unsigned int* Wbf = (unsigned int*)((char*)d_ws + (size_t)N * D_FEAT * sizeof(float));

  // Zero the accumulator region (graph-capture-safe memset node).
  hipMemsetAsync(hn, 0, (size_t)N * D_FEAT * sizeof(float), stream);

  // Convert W to packed bf16 once per launch (independent of scatter).
  kgat_prep_w<<<(D_FEAT * D_FEAT / 2 + 255) / 256, 256, 0, stream>>>(W, Wbf);

  // Scatter: one wave32 per edge.
  long sthreads = (long)E * 32;
  kgat_scatter<<<(int)((sthreads + 255) / 256), 256, 0, stream>>>(
      nfeat, esrc, edst, ew, hn, E);

  // Fused bi-interaction GEMM: one wave per 16-row tile, 8 waves/block.
  int ntiles = (N + 15) / 16;
  kgat_gemm<<<(ntiles + 7) / 8, 256, 0, stream>>>(nfeat, hn, Wbf, out, N);
}
